// BuddyPool_42537356100368
// MI455X (gfx1250) — matrix operand order repaired and checked by reading, not compiled, not run on hardware
//
#include <hip/hip_runtime.h>

// BuddyPool for MI455X (gfx1250, wave32).
// Phase 0: prep — cue rows pre-split into f16 hi/lo A-fragments in exact WMMA
//          lane layout, plus cue inverse L2 norms (removes all cue work from
//          the hot loop).
// Phase 1: fused normalize+GEMM via V_WMMA_F32_16X16X32_F16, hi/lo split for
//          ~fp32-accurate cosine sims; 2 n-tiles per wave so each A fragment
//          feeds 2 B fragments. HBM-bound: patches read exactly once.
//          Sum-of-squares uses 8 independent partial accumulators per tile to
//          avoid a serial v_add_f32 dependency chain in the hot loop.
// Phase 2: per-(b,k) top-9 selection in LDS + gather of normalized rows + mean.

typedef __attribute__((ext_vector_type(16))) _Float16 v16h;
typedef __attribute__((ext_vector_type(8)))  float    v8f;

#define BB   16
#define KQ   5
#define NP   4096
#define DD   1024
#define TOPK 9
#define EPSN 1e-12f
#define KB32 (DD / 32)            // 32 K-blocks per row

// A-fragment element e of lane (h, m) holds cue[m][kb + 16h + e].
__device__ __forceinline__ int a_k_of(int e, int h) {
    return 16 * h + e;
}

// ---------------------------------------------------------------------------
// Kernel 0: build A fragments (hi/lo) + cue inverse norms.
// One block per batch b; wave w handles K-block w (32 waves = 1024 threads).
// afrag layout: [b][kb32][lane] -> 32 halves (ahi[0..15], alo[0..15]).
// ---------------------------------------------------------------------------
__global__ __launch_bounds__(1024) void
prep_cue_kernel(const float* __restrict__ cue,
                _Float16* __restrict__ afrag,   // BB*KB32*32*32 halves
                float* __restrict__ cueInv)     // BB*KQ
{
    __shared__ float s_sq[16];

    const int b    = blockIdx.x;
    const int tid  = threadIdx.x;
    const int kb32 = tid >> 5;
    const int lane = tid & 31;
    const int h    = lane >> 4;
    const int lm   = lane & 15;
    const int kb   = kb32 * 32;

    if (tid < 16) s_sq[tid] = 0.f;
    __syncthreads();

    float ca[16];
    if (lm < KQ) {
        const float* crow = cue + ((size_t)b * KQ + lm) * DD;
        #pragma unroll
        for (int e = 0; e < 16; ++e) ca[e] = crow[kb + a_k_of(e, h)];
    } else {
        #pragma unroll
        for (int e = 0; e < 16; ++e) ca[e] = 0.f;
    }

    float sq = 0.f;
    v16h ahi, alo;
    #pragma unroll
    for (int e = 0; e < 16; ++e) {
        sq += ca[e] * ca[e];
        _Float16 xh = (_Float16)ca[e];
        ahi[e] = xh;
        alo[e] = (_Float16)(ca[e] - (float)xh);
    }

    _Float16* dst = afrag + (((size_t)b * KB32 + kb32) * 32 + lane) * 32;
    *(v16h*)(dst)      = ahi;
    *(v16h*)(dst + 16) = alo;

    if (lm < KQ) atomicAdd(&s_sq[lm], sq);
    __syncthreads();
    if (tid < KQ) cueInv[b * KQ + tid] = 1.f / fmaxf(sqrtf(s_sq[tid]), EPSN);
}

// ---------------------------------------------------------------------------
// Kernel 1: sims[b][k][n] = cosine(cue[b,k], patches[b,n]).
// One wave per (b, 32-wide n-tile): two 16-wide sub-tiles share each A frag.
// ---------------------------------------------------------------------------
__global__ __launch_bounds__(128) void
sims_wmma_kernel(const _Float16* __restrict__ afrag,
                 const float* __restrict__ cueInv,
                 const float* __restrict__ patches,
                 float* __restrict__ sims)
{
    const int wave = blockIdx.x * (blockDim.x >> 5) + (threadIdx.x >> 5);
    const int lane = threadIdx.x & 31;
    const int h    = lane >> 4;       // lane half
    const int lm   = lane & 15;       // N index within sub-tile (and C col)

    const int b    = wave >> 7;       // 128 32-wide tiles per batch
    const int n0   = (wave & 127) * 32;

    // Patch rows feeding the two B fragments (K = kb + 16h + e, contiguous).
    const float* prow0 = patches + ((size_t)b * NP + n0 + lm) * DD + h * 16;
    const float* prow1 = prow0 + (size_t)16 * DD;
    const _Float16* ap = afrag + ((size_t)b * KB32 * 32 + lane) * 32;

    v8f acc0 = {}, acc1 = {};
    // 8 independent sum-of-squares partials per sub-tile -> v_pk_fma_f32 pairs
    // with no serial dependency chain inside the K-loop.
    float sq0[8], sq1[8];
    #pragma unroll
    for (int i = 0; i < 8; ++i) { sq0[i] = 0.f; sq1[i] = 0.f; }

    for (int kb = 0; kb < DD; kb += 32) {
        // ---- A fragment: pre-swizzled hi/lo halves (L2-resident, 1 MB) ----
        v16h ahi = *(const v16h*)(ap);
        v16h alo = *(const v16h*)(ap + 16);
        ap += (size_t)32 * 32;        // next K-block

        // ---- sub-tile 0 ----
        {
            float pb[16];
            #pragma unroll
            for (int q = 0; q < 4; ++q) {
                float4 t = *(const float4*)(prow0 + kb + 4 * q);
                pb[4*q+0] = t.x; pb[4*q+1] = t.y; pb[4*q+2] = t.z; pb[4*q+3] = t.w;
            }
            v16h bhi, blo;
            #pragma unroll
            for (int e = 0; e < 16; ++e) {
                sq0[e & 7] += pb[e] * pb[e];
                _Float16 yh = (_Float16)pb[e];
                bhi[e] = yh;
                blo[e] = (_Float16)(pb[e] - (float)yh);
            }
            acc0 = __builtin_amdgcn_wmma_f32_16x16x32_f16(false, ahi, false, bhi,
                                                          (short)0, acc0, false, false);
            acc0 = __builtin_amdgcn_wmma_f32_16x16x32_f16(false, ahi, false, blo,
                                                          (short)0, acc0, false, false);
            acc0 = __builtin_amdgcn_wmma_f32_16x16x32_f16(false, alo, false, bhi,
                                                          (short)0, acc0, false, false);
        }
        // ---- sub-tile 1 ----
        {
            float pb[16];
            #pragma unroll
            for (int q = 0; q < 4; ++q) {
                float4 t = *(const float4*)(prow1 + kb + 4 * q);
                pb[4*q+0] = t.x; pb[4*q+1] = t.y; pb[4*q+2] = t.z; pb[4*q+3] = t.w;
            }
            v16h bhi, blo;
            #pragma unroll
            for (int e = 0; e < 16; ++e) {
                sq1[e & 7] += pb[e] * pb[e];
                _Float16 yh = (_Float16)pb[e];
                bhi[e] = yh;
                blo[e] = (_Float16)(pb[e] - (float)yh);
            }
            acc1 = __builtin_amdgcn_wmma_f32_16x16x32_f16(false, ahi, false, bhi,
                                                          (short)0, acc1, false, false);
            acc1 = __builtin_amdgcn_wmma_f32_16x16x32_f16(false, ahi, false, blo,
                                                          (short)0, acc1, false, false);
            acc1 = __builtin_amdgcn_wmma_f32_16x16x32_f16(false, alo, false, bhi,
                                                          (short)0, acc1, false, false);
        }
    }

    // Fold the 8 partials (tree), then fold lane halves (wave32 xor-16).
    float patSq0 = 0.f, patSq1 = 0.f;
    #pragma unroll
    for (int s = 4; s > 0; s >>= 1) {
        #pragma unroll
        for (int i = 0; i < s; ++i) {
            sq0[i] += sq0[i + s];
            sq1[i] += sq1[i + s];
        }
    }
    patSq0 = sq0[0];
    patSq1 = sq1[0];
    float patInv0 = 1.f / fmaxf(sqrtf(patSq0 + __shfl_xor(patSq0, 16, 32)), EPSN);
    float patInv1 = 1.f / fmaxf(sqrtf(patSq1 + __shfl_xor(patSq1, 16, 32)), EPSN);

    // C layout: acc[r] @ lane(h,lm) = D[M = r + 8h][N = lm]; rows 0..4 in h==0.
    if (h == 0) {
        #pragma unroll
        for (int r = 0; r < KQ; ++r) {
            float ci = cueInv[b * KQ + r];
            sims[((size_t)b * KQ + r) * NP + n0 + lm]      = acc0[r] * ci * patInv0;
            sims[((size_t)b * KQ + r) * NP + n0 + 16 + lm] = acc1[r] * ci * patInv1;
        }
    }
}

// ---------------------------------------------------------------------------
// Kernel 2: per (b,k): top-9 over sims row (lower-index tie-break, matching
// jax.lax.top_k), then mean of the 9 L2-normalized patch rows.
// One 256-thread block per (b,k) pair (80 blocks total).
// ---------------------------------------------------------------------------
__global__ __launch_bounds__(256) void
topk_gather_kernel(const float* __restrict__ sims,
                   const float* __restrict__ patches,
                   float* __restrict__ out)
{
    __shared__ float s_val[NP];      // 16 KB sims row
    __shared__ float r_val[256];
    __shared__ int   r_idx[256];
    __shared__ int   s_top[TOPK];

    const int bk  = blockIdx.x;
    const int b   = bk / KQ;
    const int tid = threadIdx.x;

    const float* srow = sims + (size_t)bk * NP;
    for (int i = tid; i < NP; i += 256) s_val[i] = srow[i];
    __syncthreads();

    // ---- 9 rounds of masked argmax ----
    for (int t = 0; t < TOPK; ++t) {
        float bv = -__builtin_inff();
        int   bi = 0x7fffffff;
        for (int i = tid; i < NP; i += 256) {
            float v = s_val[i];
            if (v > bv || (v == bv && i < bi)) { bv = v; bi = i; }
        }
        r_val[tid] = bv;
        r_idx[tid] = bi;
        __syncthreads();
        for (int s = 128; s > 0; s >>= 1) {
            if (tid < s) {
                float ov = r_val[tid + s];
                int   oi = r_idx[tid + s];
                if (ov > r_val[tid] || (ov == r_val[tid] && oi < r_idx[tid])) {
                    r_val[tid] = ov;
                    r_idx[tid] = oi;
                }
            }
            __syncthreads();
        }
        if (tid == 0) {
            s_top[t] = r_idx[0];
            s_val[r_idx[0]] = -__builtin_inff();   // mask winner
        }
        __syncthreads();
    }

    // ---- gather the 9 rows, normalize on the fly, accumulate mean ----
    float4 accv = make_float4(0.f, 0.f, 0.f, 0.f);
    for (int t = 0; t < TOPK; ++t) {
        const float* row = patches + ((size_t)b * NP + s_top[t]) * DD;
        float4 xv = *(const float4*)(row + 4 * tid);
        float  ss = xv.x * xv.x + xv.y * xv.y + xv.z * xv.z + xv.w * xv.w;
        r_val[tid] = ss;
        __syncthreads();
        for (int s = 128; s > 0; s >>= 1) {
            if (tid < s) r_val[tid] += r_val[tid + s];
            __syncthreads();
        }
        float inv = 1.f / fmaxf(sqrtf(r_val[0]), EPSN);
        accv.x += xv.x * inv;
        accv.y += xv.y * inv;
        accv.z += xv.z * inv;
        accv.w += xv.w * inv;
        __syncthreads();   // r_val reused next round
    }

    const float sc = 1.f / (float)TOPK;
    float4 o = make_float4(accv.x * sc, accv.y * sc, accv.z * sc, accv.w * sc);
    *(float4*)(out + (size_t)bk * DD + 4 * tid) = o;
}

// ---------------------------------------------------------------------------
extern "C" void kernel_launch(void* const* d_in, const int* in_sizes, int n_in,
                              void* d_out, int out_size, void* d_ws, size_t ws_size,
                              hipStream_t stream)
{
    (void)in_sizes; (void)n_in; (void)out_size; (void)ws_size;
    const float* cue     = (const float*)d_in[0];   // (16, 5, 1024) f32
    const float* patches = (const float*)d_in[1];   // (16, 4096, 1024) f32
    float*       out     = (float*)d_out;           // (16, 5, 1024) f32

    // Workspace layout (all 256B aligned):
    //   sims   : BB*KQ*NP f32            = 1,310,720 B
    //   afrag  : BB*KB32*32*32 f16       = 1,048,576 B
    //   cueInv : BB*KQ f32               = 320 B
    char* ws = (char*)d_ws;
    float*    sims    = (float*)ws;
    _Float16* afrag   = (_Float16*)(ws + ((size_t)BB * KQ * NP * 4));
    float*    cueInvP = (float*)(ws + ((size_t)BB * KQ * NP * 4)
                                    + ((size_t)BB * KB32 * 32 * 32 * 2));

    // Phase 0: one block per batch, wave w builds K-block w's A fragment.
    prep_cue_kernel<<<dim3(BB), dim3(32 * KB32), 0, stream>>>(cue, afrag, cueInvP);

    // Phase 1: B*(N/32) = 2048 waves, 4 waves (128 threads) per block.
    sims_wmma_kernel<<<dim3(BB * (NP / 32) / 4), dim3(128), 0, stream>>>(
        afrag, cueInvP, patches, sims);

    // Phase 2: one block per (b,k) pair.
    topk_gather_kernel<<<dim3(BB * KQ), dim3(256), 0, stream>>>(
        sims, patches, out);
}
